// PointNetPlusEncoder_86577950753104
// MI455X (gfx1250) — matrix operand order repaired and checked by reading, compiled, not run on hardware
//
#include <hip/hip_runtime.h>
#include <cstdint>
#include <cstddef>

typedef __attribute__((ext_vector_type(16))) _Float16 v16h;
typedef __attribute__((ext_vector_type(8)))  _Float16 v8h;
typedef __attribute__((ext_vector_type(8)))  float    v8f;

// ---------------------------------------------------------------------------
// CDNA5 wave32 WMMA fragment helpers for V_WMMA_F32_16X16X32_F16.
//
// A 16x32 f16 (ISA 7.12.2): lanes 0-15 hold rows M=0..15 (lanes 16-31 same
// rows, K offset +8): halves 0..7 -> K=kb+koff..+7, halves 8..15 -> +16.
// B 32x16 f16: lane holds column N = n0 + (lane&15); halves use the same K
// order as A (0..7 -> K=kb+koff..+7, 8..15 -> +16), strided reads of Wt[Cpad][O].
// C/D 16x16 f32: VGPR r: M = r + 8*(lane>=16), N = lane&15.
// ---------------------------------------------------------------------------
__device__ __forceinline__ v16h load_a_frag(const _Float16* __restrict__ src, int stride,
                                            int m0, int kb, int lane) {
  const int m    = m0 + (lane & 15);
  const int koff = (lane & 16) ? 8 : 0;
  const _Float16* p = src + (size_t)m * stride + kb + koff;
  v8h a0 = *reinterpret_cast<const v8h*>(p);
  v8h a1 = *reinterpret_cast<const v8h*>(p + 16);
  return __builtin_shufflevector(a0, a1, 0,1,2,3,4,5,6,7,8,9,10,11,12,13,14,15);
}

__device__ __forceinline__ v16h load_b_frag(const _Float16* __restrict__ wt, int O,
                                            int kb, int n0, int lane) {
  const int n    = n0 + (lane & 15);
  const int koff = (lane & 16) ? 8 : 0;
  v16h b;
#pragma unroll
  for (int s = 0; s < 8; ++s) {
    b[s]     = wt[(size_t)(kb + koff + s) * O + n];
    b[s + 8] = wt[(size_t)(kb + 16 + koff + s) * O + n];
  }
  return b;
}

struct acc2 { v8f c0, c1; };

// 16x32 output tile per wave: one A fragment feeds two back-to-back WMMAs
// (A-reuse; B fragments are strided column reads of the transposed weights).
__device__ __forceinline__ acc2 wmma_tile2(const _Float16* __restrict__ src, int sstride, int m0,
                                           const _Float16* __restrict__ wt, int O, int n0,
                                           int Cpad, int lane) {
  acc2 r; r.c0 = v8f{}; r.c1 = v8f{};
  for (int kb = 0; kb < Cpad; kb += 32) {
    if (kb + 32 < Cpad)  // stream next weight slice through L2 (global_prefetch_b8)
      __builtin_prefetch(wt + (size_t)(kb + 32 + (lane & 15)) * O + n0, 0, 1);
    v16h a  = load_a_frag(src, sstride, m0, kb, lane);
    v16h b0 = load_b_frag(wt, O, kb, n0, lane);
    v16h b1 = load_b_frag(wt, O, kb, n0 + 16, lane);
    r.c0 = __builtin_amdgcn_wmma_f32_16x16x32_f16(false, a, false, b0, (short)0, r.c0, false, false);
    r.c1 = __builtin_amdgcn_wmma_f32_16x16x32_f16(false, a, false, b1, (short)0, r.c1, false, false);
  }
  return r;
}

// ---------------------------------------------------------------------------
// Weight convert + transpose: W[O][C] f32 -> Wt[Cpad][O] f16, zero-padded K.
// cpts != 0 permutes input channels to [pts(cpts), xyz(3)] so the SA2 feature
// tile can be staged with 16B-aligned async LDS copies (pts first).
// ---------------------------------------------------------------------------
__global__ void wconvert_kernel(const float* __restrict__ w, _Float16* __restrict__ wt,
                                int O, int C, int Cpad, int cpts) {
  int t = blockIdx.x * blockDim.x + threadIdx.x;
  if (t >= Cpad * O) return;
  int k = t / O, o = t - k * O;
  float v = 0.0f;
  if (k < C) {
    int src = (cpts == 0) ? k : ((k < cpts) ? (k + 3) : (k - cpts));
    v = w[o * C + src];
  }
  wt[t] = (_Float16)v;
}

// ---------------------------------------------------------------------------
// Farthest point sampling: one workgroup per batch. Running min distance in
// registers; LDS tree argmax per round (ties -> lowest idx, jnp.argmax).
// ---------------------------------------------------------------------------
template<int NPTS, int NPOINT, int THREADS>
__global__ __launch_bounds__(THREADS) void fps_kernel(const float* __restrict__ xyz,
                                                      int* __restrict__ out_idx,
                                                      float* __restrict__ out_xyz) {
  const int b = blockIdx.x;
  const int tid = threadIdx.x;
  const float* __restrict__ p = xyz + (size_t)b * NPTS * 3;
  constexpr int PTS = NPTS / THREADS;
  float dist[PTS];
#pragma unroll
  for (int i = 0; i < PTS; ++i) dist[i] = 1e10f;
  __shared__ float fp[3];
  __shared__ float redv[THREADS];
  __shared__ int   redi[THREADS];
  __shared__ int   sfar;
  int far = 0;
  for (int it = 0; it < NPOINT; ++it) {
    if (tid == 0) {
      out_idx[b * NPOINT + it] = far;
      float fx = p[far * 3 + 0], fy = p[far * 3 + 1], fz = p[far * 3 + 2];
      fp[0] = fx; fp[1] = fy; fp[2] = fz;
      out_xyz[((size_t)b * NPOINT + it) * 3 + 0] = fx;
      out_xyz[((size_t)b * NPOINT + it) * 3 + 1] = fy;
      out_xyz[((size_t)b * NPOINT + it) * 3 + 2] = fz;
    }
    __syncthreads();
    const float fx = fp[0], fy = fp[1], fz = fp[2];
    float best = -1.0f; int bi = 0;
#pragma unroll
    for (int i = 0; i < PTS; ++i) {
      const int j = tid + i * THREADS;
      float dx = p[j * 3 + 0] - fx, dy = p[j * 3 + 1] - fy, dz = p[j * 3 + 2] - fz;
      float d = fminf(dist[i], dx * dx + dy * dy + dz * dz);
      dist[i] = d;
      if (d > best) { best = d; bi = j; }
    }
    redv[tid] = best; redi[tid] = bi;
    __syncthreads();
    for (int s = THREADS / 2; s > 0; s >>= 1) {
      if (tid < s) {
        float ov = redv[tid + s]; int oi = redi[tid + s];
        float mv = redv[tid];     int mi = redi[tid];
        if (ov > mv || (ov == mv && oi < mi)) { redv[tid] = ov; redi[tid] = oi; }
      }
      __syncthreads();
    }
    if (tid == 0) sfar = redi[0];
    __syncthreads();
    far = sfar;
  }
}

// ---------------------------------------------------------------------------
// Ball query: one thread per centroid; first K in-radius indices ascending,
// pad misses with first hit (== reference's sort-then-pad semantics).
// ---------------------------------------------------------------------------
__global__ void ball_query_kernel(const float* __restrict__ xyz, const float* __restrict__ new_xyz,
                                  int* __restrict__ idx, int BS, int n, int S, int K, float r2) {
  int t = blockIdx.x * blockDim.x + threadIdx.x;
  if (t >= BS) return;
  int b = t / S;
  const float* __restrict__ p = xyz + (size_t)b * n * 3;
  float cx = new_xyz[(size_t)t * 3 + 0], cy = new_xyz[(size_t)t * 3 + 1], cz = new_xyz[(size_t)t * 3 + 2];
  int cnt = 0, first = 0;
  for (int j = 0; j < n && cnt < K; ++j) {
    float dx = p[j * 3 + 0] - cx, dy = p[j * 3 + 1] - cy, dz = p[j * 3 + 2] - cz;
    if (dx * dx + dy * dy + dz * dz <= r2) {
      if (cnt == 0) first = j;
      idx[(size_t)t * K + cnt] = j;
      ++cnt;
    }
  }
  for (int k = cnt; k < K; ++k) idx[(size_t)t * K + k] = first;
}

// ---------------------------------------------------------------------------
// Fused set-abstraction stage: gather features into LDS (async for the pts
// channels when present), 3 WMMA GEMM+bias+ReLU layers ping-ponging two LDS
// buffers, max-pool over nsample. Only pooled [B,S,O2] f16 leaves the WGP.
// Channel layout: CF==3 -> [xyz,pad]; CF>3 -> [pts(CF-3), xyz(3), pad]
// (weights row-permuted to match in wconvert_kernel).
// ---------------------------------------------------------------------------
template<int KN, int CF, int C0PAD, int O0, int O1, int O2, int MWG, int COLS_A, int COLS_B>
__global__ __launch_bounds__(256) void sa_fused_kernel(
    const float* __restrict__ xyz, int n,
    const _Float16* __restrict__ pts,
    const float* __restrict__ new_xyz, int S,
    const int* __restrict__ idx,
    const _Float16* __restrict__ wt0, const float* __restrict__ b0,
    const _Float16* __restrict__ wt1, const float* __restrict__ b1,
    const _Float16* __restrict__ wt2, const float* __restrict__ b2,
    _Float16* __restrict__ pooled) {
  __shared__ __attribute__((aligned(32))) _Float16 bufA[MWG * COLS_A];
  __shared__ __attribute__((aligned(32))) _Float16 bufB[MWG * COLS_B];
  const int tid  = threadIdx.x;
  const int lane = tid & 31;
  const int wave = tid >> 5;
  constexpr int GW = MWG / KN;
  const int g0 = blockIdx.x * GW;

  if (CF > 3) {
    // ---- async stage: copy each neighbor's pts row (16B chunks) into LDS ---
    constexpr int CPTS   = (CF > 3) ? (CF - 3) : 8;   // halves per row (mult of 8)
    constexpr int CHUNKS = CPTS / 8;                  // 16B chunks per row
    for (int e = tid; e < MWG * CHUNKS; e += 256) {
      const int row = e / CHUNKS, part = e - row * CHUNKS;
      const int g = row / KN, k = row - g * KN;
      const int gg = g0 + g;
      const int b = gg / S;
      const int j = idx[(size_t)gg * KN + k];
      const _Float16* gp = pts + ((size_t)b * n + j) * CPTS + part * 8;
      unsigned la = (unsigned)(uintptr_t)(&bufA[row * COLS_A + part * 8]);
      asm volatile("global_load_async_to_lds_b128 %0, %1, off"
                   :: "v"(la), "v"(gp) : "memory");
    }
    // ---- VALU stage: centered xyz at channels [CPTS, CPTS+3), zero pad -----
    for (int e = tid; e < MWG * (C0PAD - CPTS); e += 256) {
      const int row = e / (C0PAD - CPTS), cc = e - row * (C0PAD - CPTS);
      const int g = row / KN, k = row - g * KN;
      const int gg = g0 + g;
      const int b = gg / S;
      const int j = idx[(size_t)gg * KN + k];
      float v = 0.0f;
      if (cc < 3) v = xyz[((size_t)b * n + j) * 3 + cc] - new_xyz[(size_t)gg * 3 + cc];
      bufA[row * COLS_A + CPTS + cc] = (_Float16)v;
    }
    asm volatile("s_wait_asynccnt 0x0" ::: "memory");
  } else {
    for (int e = tid; e < MWG * C0PAD; e += 256) {
      const int row = e / C0PAD, c = e - row * C0PAD;
      const int g = row / KN, k = row - g * KN;
      const int gg = g0 + g;
      const int b = gg / S;
      const int j = idx[(size_t)gg * KN + k];
      float v = 0.0f;
      if (c < 3) v = xyz[((size_t)b * n + j) * 3 + c] - new_xyz[(size_t)gg * 3 + c];
      bufA[row * COLS_A + c] = (_Float16)v;
    }
  }
  __syncthreads();

  auto layer = [&](const _Float16* src, int sstride, int Cpad,
                   _Float16* dst, int dstride, int O,
                   const _Float16* wt, const float* bias) {
    const int TM = MWG / 16, T = TM * (O / 32);
    for (int t = wave; t < T; t += 8) {          // wave-uniform: EXEC stays full
      const int mt = t % TM, nt = t / TM;
      const int n0 = nt * 32;
      acc2 a2 = wmma_tile2(src, sstride, mt * 16, wt, O, n0, Cpad, lane);
      const int nn = n0 + (lane & 15);
      const float bv0 = bias[nn], bv1 = bias[nn + 16];
      const int mbase = mt * 16 + ((lane & 16) ? 8 : 0);
#pragma unroll
      for (int r = 0; r < 8; ++r) {
        dst[(mbase + r) * dstride + nn]      = (_Float16)fmaxf(a2.c0[r] + bv0, 0.0f);
        dst[(mbase + r) * dstride + nn + 16] = (_Float16)fmaxf(a2.c1[r] + bv1, 0.0f);
      }
    }
  };
  layer(bufA, COLS_A, C0PAD, bufB, COLS_B, O0, wt0, b0);
  __syncthreads();
  layer(bufB, COLS_B, O0, bufA, COLS_A, O1, wt1, b1);
  __syncthreads();
  layer(bufA, COLS_A, O1, bufB, COLS_B, O2, wt2, b2);
  __syncthreads();

  // Max-pool over the KN neighbors of each group.
  for (int e = tid; e < GW * O2; e += 256) {
    const int g = e / O2, o = e - g * O2;
    float m = -1e30f;
    for (int k = 0; k < KN; ++k)
      m = fmaxf(m, (float)bufB[(g * KN + k) * COLS_B + o]);
    pooled[(size_t)(g0 + g) * O2 + o] = (_Float16)m;
  }
}

// ---------------------------------------------------------------------------
// Generic WMMA GEMM (global->global), one 16x32 tile per wave (A-reuse).
// ---------------------------------------------------------------------------
template<bool RELU, bool F32OUT>
__global__ __launch_bounds__(256) void gemm_kernel(
    const _Float16* __restrict__ A, int lda,
    const _Float16* __restrict__ wt, const float* __restrict__ bias,
    void* __restrict__ out, int M, int Cpad, int O) {
  const int lane = threadIdx.x & 31;
  const int wave = threadIdx.x >> 5;
  const int tile = blockIdx.x * 8 + wave;
  const int TM = M / 16;
  if (tile >= TM * (O / 32)) return;       // wave-uniform exit
  const int mt = tile % TM, nt = tile / TM;
  const int n0 = nt * 32;
  acc2 a2 = wmma_tile2(A, lda, mt * 16, wt, O, n0, Cpad, lane);
  const int nn = n0 + (lane & 15);
  const float bv0 = bias[nn], bv1 = bias[nn + 16];
  const int mbase = mt * 16 + ((lane & 16) ? 8 : 0);
#pragma unroll
  for (int r = 0; r < 8; ++r) {
    float v0 = a2.c0[r] + bv0;
    float v1 = a2.c1[r] + bv1;
    if (RELU) { v0 = fmaxf(v0, 0.0f); v1 = fmaxf(v1, 0.0f); }
    if (F32OUT) {
      ((float*)out)[(size_t)(mbase + r) * O + nn]      = v0;
      ((float*)out)[(size_t)(mbase + r) * O + nn + 16] = v1;
    } else {
      ((_Float16*)out)[(size_t)(mbase + r) * O + nn]      = (_Float16)v0;
      ((_Float16*)out)[(size_t)(mbase + r) * O + nn + 16] = (_Float16)v1;
    }
  }
}

// SA3 group_all feature assembly: [16*128, 288] = concat(l2_xyz, l2_pts), pad.
__global__ void feat3_kernel(const float* __restrict__ l2xyz, const _Float16* __restrict__ l2pts,
                             _Float16* __restrict__ feat) {
  int t = blockIdx.x * blockDim.x + threadIdx.x;
  if (t >= 2048 * 288) return;
  int row = t / 288, c = t - row * 288;
  float v = 0.0f;
  if (c < 3)        v = l2xyz[row * 3 + c];
  else if (c < 259) v = (float)l2pts[(size_t)row * 256 + (c - 3)];
  feat[t] = (_Float16)v;
}

// SA3 max-pool over the 128 group_all rows per batch.
__global__ void maxpool3_kernel(const _Float16* __restrict__ t2, _Float16* __restrict__ g) {
  int t = blockIdx.x * blockDim.x + threadIdx.x;
  if (t >= 16 * 1024) return;
  int b = t >> 10, o = t & 1023;
  float m = -1e30f;
  for (int k = 0; k < 128; ++k)
    m = fmaxf(m, (float)t2[((size_t)b * 128 + k) * 1024 + o]);
  g[t] = (_Float16)m;
}

// ---------------------------------------------------------------------------
extern "C" void kernel_launch(void* const* d_in, const int* in_sizes, int n_in,
                              void* d_out, int out_size, void* d_ws, size_t ws_size,
                              hipStream_t stream) {
  const float* x = (const float*)d_in[0];
  const float* w_raw[10]; const float* b_raw[10];
  for (int i = 0; i < 9; ++i) { w_raw[i] = (const float*)d_in[1 + 2 * i]; b_raw[i] = (const float*)d_in[2 + 2 * i]; }
  w_raw[9] = (const float*)d_in[19]; b_raw[9] = (const float*)d_in[20];

  static const int O_[10]    = {64, 64, 128, 128, 128, 256, 256, 512, 1024, 1024};
  static const int C_[10]    = {3,  64, 64,  131, 128, 128, 259, 256, 512,  1024};
  static const int CPAD_[10] = {32, 64, 64,  160, 128, 128, 288, 256, 512,  1024};
  static const int PERM_[10] = {0,  0,  0,   128, 0,   0,   0,   0,   0,    0};

  char* p = (char*)d_ws;
  auto carve = [&](size_t bytes) -> void* {
    void* r = (void*)p;
    p += (bytes + 255) & ~(size_t)255;
    return r;
  };
  _Float16* wt[10];
  for (int i = 0; i < 10; ++i) wt[i] = (_Float16*)carve((size_t)CPAD_[i] * O_[i] * sizeof(_Float16));
  int*      fps_idx1 = (int*)carve((size_t)16 * 512 * sizeof(int));
  float*    new_xyz1 = (float*)carve((size_t)16 * 512 * 3 * sizeof(float));
  int*      idx1     = (int*)carve((size_t)16 * 512 * 32 * sizeof(int));
  _Float16* l1_pts   = (_Float16*)carve((size_t)16 * 512 * 128 * sizeof(_Float16));
  int*      fps_idx2 = (int*)carve((size_t)16 * 128 * sizeof(int));
  float*    new_xyz2 = (float*)carve((size_t)16 * 128 * 3 * sizeof(float));
  int*      idx2     = (int*)carve((size_t)16 * 128 * 64 * sizeof(int));
  _Float16* l2_pts   = (_Float16*)carve((size_t)16 * 128 * 256 * sizeof(_Float16));
  _Float16* feat3    = (_Float16*)carve((size_t)2048 * 288 * sizeof(_Float16));
  _Float16* t0       = (_Float16*)carve((size_t)2048 * 256 * sizeof(_Float16));
  _Float16* t1       = (_Float16*)carve((size_t)2048 * 512 * sizeof(_Float16));
  _Float16* t2       = (_Float16*)carve((size_t)2048 * 1024 * sizeof(_Float16));
  _Float16* gbuf     = (_Float16*)carve((size_t)16 * 1024 * sizeof(_Float16));

  for (int i = 0; i < 10; ++i) {
    int total = CPAD_[i] * O_[i];
    wconvert_kernel<<<(total + 255) / 256, 256, 0, stream>>>(w_raw[i], wt[i], O_[i], C_[i], CPAD_[i], PERM_[i]);
  }

  // SA1: FPS 4096->512, ball r=0.2 K=32, MLP 3->64->64->128, pool.
  fps_kernel<4096, 512, 1024><<<16, 1024, 0, stream>>>(x, fps_idx1, new_xyz1);
  ball_query_kernel<<<(16 * 512 + 255) / 256, 256, 0, stream>>>(x, new_xyz1, idx1, 16 * 512, 4096, 512, 32, 0.04f);
  sa_fused_kernel<32, 3, 32, 64, 64, 128, 128, 64, 128><<<2048, 256, 0, stream>>>(
      x, 4096, (const _Float16*)nullptr, new_xyz1, 512, idx1,
      wt[0], b_raw[0], wt[1], b_raw[1], wt[2], b_raw[2], l1_pts);

  // SA2: FPS 512->128, ball r=0.4 K=64, MLP 131->128->128->256, pool.
  fps_kernel<512, 128, 512><<<16, 512, 0, stream>>>(new_xyz1, fps_idx2, new_xyz2);
  ball_query_kernel<<<(16 * 128 + 255) / 256, 256, 0, stream>>>(new_xyz1, new_xyz2, idx2, 16 * 128, 512, 128, 64, 0.16f);
  sa_fused_kernel<64, 131, 160, 128, 128, 256, 64, 160, 256><<<2048, 256, 0, stream>>>(
      new_xyz1, 512, l1_pts, new_xyz2, 128, idx2,
      wt[3], b_raw[3], wt[4], b_raw[4], wt[5], b_raw[5], l2_pts);

  // SA3 (group_all): 259->256->512->1024, pool over 128, then FC 1024x1024.
  feat3_kernel<<<(2048 * 288 + 255) / 256, 256, 0, stream>>>(new_xyz2, l2_pts, feat3);
  gemm_kernel<true, false><<<128, 256, 0, stream>>>(feat3, 288, wt[6], b_raw[6], t0, 2048, 288, 256);
  gemm_kernel<true, false><<<256, 256, 0, stream>>>(t0, 256, wt[7], b_raw[7], t1, 2048, 256, 512);
  gemm_kernel<true, false><<<512, 256, 0, stream>>>(t1, 512, wt[8], b_raw[8], t2, 2048, 512, 1024);
  maxpool3_kernel<<<64, 256, 0, stream>>>(t2, gbuf);
  gemm_kernel<false, true><<<4, 256, 0, stream>>>(gbuf, 1024, wt[9], b_raw[9], (void*)d_out, 16, 1024, 1024);

  (void)in_sizes; (void)n_in; (void)out_size; (void)ws_size;
  (void)fps_idx1; (void)fps_idx2;
}
